// MultiheadedAttention_47691316854802
// MI455X (gfx1250) — matrix operand order repaired and checked by reading, not compiled, not run on hardware
//
#include <hip/hip_runtime.h>

// ---------------------------------------------------------------------------
// Multi-head "attention" (softmax discarded in reference => pure matmul chain)
// out = ((X Wq + bq)(X Wk + bk)^T / 8)(X Wv + bv) Wo + bo, per head
// Reassociated per head:  q @ (k^T v) / 8   (exact matmul associativity)
// bf16 WMMA (v_wmma_f32_16x16x32_bf16), fp32 accumulate, fp32 epilogue.
// ---------------------------------------------------------------------------

static constexpr int Bdim = 2;
static constexpr int Sdim = 2048;
static constexpr int Ddim = 1024;
static constexpr int Hdim = 16;

typedef __attribute__((ext_vector_type(16))) __bf16 v16bf;
typedef __attribute__((ext_vector_type(8)))  float  v8f;

struct alignas(16) Q4 { unsigned x, y, z, w; };
struct Q8 { Q4 lo, hi; };

// A-fragment (16x32 bf16): elems 0..15 <- p[0..15] (16 contiguous K values)
__device__ __forceinline__ v16bf load_fragA(const __bf16* p) {
  Q8 u;
  u.lo = *reinterpret_cast<const Q4*>(p);
  u.hi = *reinterpret_cast<const Q4*>(p + 8);
  return __builtin_bit_cast(v16bf, u);
}
// B-fragment (32x16 bf16): lane holds 16 contiguous K values -> 32 contiguous bytes
__device__ __forceinline__ v16bf load_fragB(const __bf16* p) {
  Q8 u;
  u.lo = *reinterpret_cast<const Q4*>(p);
  u.hi = *reinterpret_cast<const Q4*>(p + 8);
  return __builtin_bit_cast(v16bf, u);
}

__device__ __forceinline__ v8f wmma_bf16(v16bf a, v16bf b, v8f c) {
  return __builtin_amdgcn_wmma_f32_16x16x32_bf16(false, a, false, b, (short)0, c,
                                                 false, false);
}

// hardware bf16 convert (v_cvt_*), replaces manual RNE bit sequence
__device__ __forceinline__ __bf16 f2bf(float f) { return static_cast<__bf16>(f); }
__device__ __forceinline__ float bf2f(__bf16 h) {
  unsigned short s = __builtin_bit_cast(unsigned short, h);
  unsigned u = ((unsigned)s) << 16;
  return __builtin_bit_cast(float, u);
}
__device__ __forceinline__ v8f vzero8() {
  v8f z;
#pragma unroll
  for (int i = 0; i < 8; ++i) z[i] = 0.0f;
  return z;
}

// ---------------------------------------------------------------------------
// fp32 -> bf16 bulk convert (vectorized, packed 8B stores)
// ---------------------------------------------------------------------------
__global__ __launch_bounds__(256) void cvt_f32_bf16(const float* __restrict__ in,
                                                    __bf16* __restrict__ out, int n) {
  int i = (blockIdx.x * 256 + threadIdx.x) * 4;
  if (i + 3 < n) {
    struct alignas(16) F4 { float a, b, c, d; };
    struct alignas(8) BF4 { __bf16 a, b, c, d; };
    F4 f = *reinterpret_cast<const F4*>(in + i);
    BF4 o{f2bf(f.a), f2bf(f.b), f2bf(f.c), f2bf(f.d)};
    *reinterpret_cast<BF4*>(out + i) = o;
  }
}

// ---------------------------------------------------------------------------
// W [K,N] fp32 -> WT [N,K] bf16 (LDS tiled transpose)
// ---------------------------------------------------------------------------
__global__ __launch_bounds__(256) void wtrans_bf16(const float* __restrict__ W,
                                                   __bf16* __restrict__ WT) {
  __shared__ float tile[32][33];
  const int tx = threadIdx.x & 31;
  const int ty = threadIdx.x >> 5;  // 0..7
  const int n0 = blockIdx.x * 32;
  const int k0 = blockIdx.y * 32;
#pragma unroll
  for (int r = 0; r < 32; r += 8)
    tile[ty + r][tx] = W[(size_t)(k0 + ty + r) * Ddim + n0 + tx];
  __syncthreads();
#pragma unroll
  for (int r = 0; r < 32; r += 8)
    WT[(size_t)(n0 + ty + r) * Ddim + k0 + tx] = f2bf(tile[tx][ty + r]);
}

// ---------------------------------------------------------------------------
// C = A @ B + bias.  A bf16 [M,K] row-major, BT bf16 [N,K] row-major.
// 256 thr = 8 waves; wave tile 32x64 (2x4 wmma), block tile 128x128.
// Register double-buffered K loop: next-iter loads issued before this-iter
// WMMAs so VMEM latency overlaps matrix math.
// ---------------------------------------------------------------------------
template <bool OUT_BF16>
__global__ __launch_bounds__(256) void gemm_bf16_tn(
    const __bf16* __restrict__ A, const __bf16* __restrict__ BT,
    const float* __restrict__ bias, float* __restrict__ Cf,
    __bf16* __restrict__ Cb, int M, int N, int K) {
  const int lane = threadIdx.x & 31;
  const int wave = threadIdx.x >> 5;
  const int lm = lane & 15;
  const int hf = lane >> 4;
  const int m0 = blockIdx.y * 128 + (wave & 3) * 32;
  const int n0 = blockIdx.x * 128 + (wave >> 2) * 64;

  v8f acc[2][4];
#pragma unroll
  for (int i = 0; i < 2; ++i)
#pragma unroll
    for (int j = 0; j < 4; ++j) acc[i][j] = vzero8();

  const __bf16* pa0 = A + (size_t)(m0 + lm) * K + hf * 16;
  const __bf16* pa1 = pa0 + (size_t)16 * K;
  const __bf16* pb0 = BT + (size_t)(n0 + 0 * 16 + lm) * K + hf * 16;
  const __bf16* pb1 = BT + (size_t)(n0 + 1 * 16 + lm) * K + hf * 16;
  const __bf16* pb2 = BT + (size_t)(n0 + 2 * 16 + lm) * K + hf * 16;
  const __bf16* pb3 = BT + (size_t)(n0 + 3 * 16 + lm) * K + hf * 16;

  v16bf a0 = load_fragA(pa0);
  v16bf a1 = load_fragA(pa1);
  v16bf b0 = load_fragB(pb0);
  v16bf b1 = load_fragB(pb1);
  v16bf b2 = load_fragB(pb2);
  v16bf b3 = load_fragB(pb3);

#pragma unroll 4
  for (int k0 = 32; k0 < K; k0 += 32) {
    // prefetch next K-step fragments (12 b128 loads in flight)
    v16bf na0 = load_fragA(pa0 + k0);
    v16bf na1 = load_fragA(pa1 + k0);
    v16bf nb0 = load_fragB(pb0 + k0);
    v16bf nb1 = load_fragB(pb1 + k0);
    v16bf nb2 = load_fragB(pb2 + k0);
    v16bf nb3 = load_fragB(pb3 + k0);

    // compute on current fragments (no dependence on in-flight loads)
    acc[0][0] = wmma_bf16(a0, b0, acc[0][0]);
    acc[1][0] = wmma_bf16(a1, b0, acc[1][0]);
    acc[0][1] = wmma_bf16(a0, b1, acc[0][1]);
    acc[1][1] = wmma_bf16(a1, b1, acc[1][1]);
    acc[0][2] = wmma_bf16(a0, b2, acc[0][2]);
    acc[1][2] = wmma_bf16(a1, b2, acc[1][2]);
    acc[0][3] = wmma_bf16(a0, b3, acc[0][3]);
    acc[1][3] = wmma_bf16(a1, b3, acc[1][3]);

    a0 = na0; a1 = na1;
    b0 = nb0; b1 = nb1; b2 = nb2; b3 = nb3;
  }
  // tail K-step
  acc[0][0] = wmma_bf16(a0, b0, acc[0][0]);
  acc[1][0] = wmma_bf16(a1, b0, acc[1][0]);
  acc[0][1] = wmma_bf16(a0, b1, acc[0][1]);
  acc[1][1] = wmma_bf16(a1, b1, acc[1][1]);
  acc[0][2] = wmma_bf16(a0, b2, acc[0][2]);
  acc[1][2] = wmma_bf16(a1, b2, acc[1][2]);
  acc[0][3] = wmma_bf16(a0, b3, acc[0][3]);
  acc[1][3] = wmma_bf16(a1, b3, acc[1][3]);

#pragma unroll
  for (int mi = 0; mi < 2; ++mi)
#pragma unroll
    for (int ni = 0; ni < 4; ++ni) {
      const int col = n0 + ni * 16 + lm;
      const float bs = bias[col];
#pragma unroll
      for (int j = 0; j < 8; ++j) {
        const int row = m0 + mi * 16 + hf * 8 + j;
        const float v = acc[mi][ni][j] + bs;
        if (OUT_BF16) Cb[(size_t)row * N + col] = f2bf(v);
        else          Cf[(size_t)row * N + col] = v;
      }
    }
}

// ---------------------------------------------------------------------------
// kv[m,n] = sum_s k[s,m] * v[s,n] per (b,h); store kvT[n*64+m] = kv*0.125 bf16
// (0.5 GFLOP total -> plain VALU with LDS staging is ample)
// ---------------------------------------------------------------------------
__global__ __launch_bounds__(256) void kv_reduce(const __bf16* __restrict__ kbf,
                                                 const __bf16* __restrict__ vbf,
                                                 __bf16* __restrict__ kvT) {
  __shared__ float kt[64 * 64];
  __shared__ float vt[64 * 64];
  const int tid = threadIdx.x;
  const int bh = blockIdx.x;
  const int b = bh >> 4;
  const int hh = bh & 15;
  const size_t base = (size_t)b * Sdim * Ddim + hh * 64;

  const int m = tid >> 2;
  const int nb = (tid & 3) * 16;
  float acc[16];
#pragma unroll
  for (int j = 0; j < 16; ++j) acc[j] = 0.0f;

  for (int s0 = 0; s0 < Sdim; s0 += 64) {
    __syncthreads();
    for (int i = tid; i < 64 * 64; i += 256) {
      const int sl = i >> 6, c = i & 63;
      const size_t g = base + (size_t)(s0 + sl) * Ddim + c;
      kt[i] = bf2f(kbf[g]);
      vt[i] = bf2f(vbf[g]);
    }
    __syncthreads();
#pragma unroll 4
    for (int s = 0; s < 64; ++s) {
      const float km = kt[s * 64 + m];
#pragma unroll
      for (int j = 0; j < 16; ++j) acc[j] += km * vt[s * 64 + nb + j];
    }
  }
  __bf16* out = kvT + (size_t)bh * 64 * 64;
#pragma unroll
  for (int j = 0; j < 16; ++j)
    out[(nb + j) * 64 + m] = f2bf(acc[j] * 0.125f);
}

// ---------------------------------------------------------------------------
// outH[b,s,h*64+n] = q[b,s,h*64+:] @ kvT_head  (K=64, WMMA, per head)
// 128 thr = 4 waves; each wave 16 rows x 64 cols. All fragments preloaded,
// then 8 WMMAs issued back-to-back.
// ---------------------------------------------------------------------------
__global__ __launch_bounds__(128) void attn_apply(const __bf16* __restrict__ qbf,
                                                  const __bf16* __restrict__ kvT,
                                                  __bf16* __restrict__ outH) {
  const int lane = threadIdx.x & 31;
  const int wave = threadIdx.x >> 5;
  const int lm = lane & 15;
  const int hf = lane >> 4;
  const int bh = blockIdx.y;
  const int b = bh >> 4;
  const int hh = bh & 15;
  const int srow = blockIdx.x * 64 + wave * 16;

  const __bf16* pa = qbf + ((size_t)b * Sdim + srow + lm) * Ddim + hh * 64 + hf * 16;
  const __bf16* pkv = kvT + (size_t)bh * 64 * 64 + hf * 16;

  // preload everything: 2 A fragments + 8 B fragments
  v16bf a0 = load_fragA(pa);
  v16bf a1 = load_fragA(pa + 32);
  v16bf bb[2][4];
#pragma unroll
  for (int ks = 0; ks < 2; ++ks)
#pragma unroll
    for (int ni = 0; ni < 4; ++ni)
      bb[ks][ni] = load_fragB(pkv + (ni * 16 + lm) * 64 + ks * 32);

  v8f acc[4];
#pragma unroll
  for (int i = 0; i < 4; ++i) acc[i] = vzero8();
#pragma unroll
  for (int ni = 0; ni < 4; ++ni) acc[ni] = wmma_bf16(a0, bb[0][ni], acc[ni]);
#pragma unroll
  for (int ni = 0; ni < 4; ++ni) acc[ni] = wmma_bf16(a1, bb[1][ni], acc[ni]);

#pragma unroll
  for (int ni = 0; ni < 4; ++ni)
#pragma unroll
    for (int j = 0; j < 8; ++j) {
      const int row = srow + hf * 8 + j;
      outH[((size_t)b * Sdim + row) * Ddim + hh * 64 + ni * 16 + lm] =
          f2bf(acc[ni][j]);
    }
}

// ---------------------------------------------------------------------------
extern "C" void kernel_launch(void* const* d_in, const int* in_sizes, int n_in,
                              void* d_out, int out_size, void* d_ws, size_t ws_size,
                              hipStream_t stream) {
  const float* Q  = (const float*)d_in[0];
  const float* K  = (const float*)d_in[1];
  const float* V  = (const float*)d_in[2];
  const float* Wq = (const float*)d_in[3];
  const float* Wk = (const float*)d_in[4];
  const float* Wv = (const float*)d_in[5];
  const float* Wo = (const float*)d_in[6];
  const float* bq = (const float*)d_in[7];
  const float* bk = (const float*)d_in[8];
  const float* bv = (const float*)d_in[9];
  const float* bo = (const float*)d_in[10];
  float* out = (float*)d_out;

  const size_t actElems = (size_t)Bdim * Sdim * Ddim;  // 4,194,304
  const size_t actB = actElems * 2;                    // bf16 bytes (8 MB)
  const size_t wB = (size_t)Ddim * Ddim * 2;           // 2 MB

  char* ws = (char*)d_ws;
  __bf16* Qbf = (__bf16*)(ws + 0 * actB);
  __bf16* Kbf = (__bf16*)(ws + 1 * actB);
  __bf16* Vbf = (__bf16*)(ws + 2 * actB);
  __bf16* qP  = (__bf16*)(ws + 3 * actB);
  __bf16* kP  = (__bf16*)(ws + 4 * actB);
  __bf16* vP  = (__bf16*)(ws + 5 * actB);
  __bf16* oH  = (__bf16*)(ws + 6 * actB);
  __bf16* WqT = (__bf16*)(ws + 7 * actB + 0 * wB);
  __bf16* WkT = (__bf16*)(ws + 7 * actB + 1 * wB);
  __bf16* WvT = (__bf16*)(ws + 7 * actB + 2 * wB);
  __bf16* WoT = (__bf16*)(ws + 7 * actB + 3 * wB);
  __bf16* kvT = (__bf16*)(ws + 7 * actB + 4 * wB);  // [B*H][64*64]

  const int n = (int)actElems;
  dim3 cvtGrid(n / (256 * 4));
  cvt_f32_bf16<<<cvtGrid, 256, 0, stream>>>(Q, Qbf, n);
  cvt_f32_bf16<<<cvtGrid, 256, 0, stream>>>(K, Kbf, n);
  cvt_f32_bf16<<<cvtGrid, 256, 0, stream>>>(V, Vbf, n);

  dim3 wtGrid(Ddim / 32, Ddim / 32);
  wtrans_bf16<<<wtGrid, 256, 0, stream>>>(Wq, WqT);
  wtrans_bf16<<<wtGrid, 256, 0, stream>>>(Wk, WkT);
  wtrans_bf16<<<wtGrid, 256, 0, stream>>>(Wv, WvT);
  wtrans_bf16<<<wtGrid, 256, 0, stream>>>(Wo, WoT);

  const int M = Bdim * Sdim;         // 4096
  dim3 gGrid(Ddim / 128, M / 128);   // (8, 32)
  gemm_bf16_tn<true><<<gGrid, 256, 0, stream>>>(Qbf, WqT, bq, nullptr, qP, M, Ddim, Ddim);
  gemm_bf16_tn<true><<<gGrid, 256, 0, stream>>>(Kbf, WkT, bk, nullptr, kP, M, Ddim, Ddim);
  gemm_bf16_tn<true><<<gGrid, 256, 0, stream>>>(Vbf, WvT, bv, nullptr, vP, M, Ddim, Ddim);

  kv_reduce<<<dim3(Bdim * Hdim), 256, 0, stream>>>(kP, vP, kvT);

  attn_apply<<<dim3(Sdim / 64, Bdim * Hdim), 128, 0, stream>>>(qP, kvT, oH);

  gemm_bf16_tn<false><<<gGrid, 256, 0, stream>>>(oH, WoT, bo, out, nullptr, M, Ddim, Ddim);

  (void)in_sizes; (void)n_in; (void)out_size; (void)ws_size;
}
